// CausalRotarySelfAttention_4544075399231
// MI455X (gfx1250) — matrix operand order repaired and checked by reading, not compiled, not run on hardware
//
#include <hip/hip_runtime.h>
#include <hip/hip_bf16.h>

#define DEV static __device__ __forceinline__

typedef __bf16 bf16_t;
typedef __attribute__((ext_vector_type(16))) __bf16 v16bf;
typedef __attribute__((ext_vector_type(8)))  float  v8f;

constexpr int BB    = 4;
constexpr int NN    = 1024;
constexpr int DD    = 1024;
constexpr int HH    = 16;
constexpr int DH    = 64;
constexpr int INNER = HH * DH;      // 1024
constexpr int E3    = 3 * INNER;    // 3072
constexpr int MROWS = BB * NN;      // 4096

// ---------- bf16 helpers (bits stored as unsigned short) ----------
DEV unsigned short f2bf(float f) {
  unsigned int u = __builtin_bit_cast(unsigned int, f);
  unsigned int r = 0x7FFFu + ((u >> 16) & 1u);
  return (unsigned short)((u + r) >> 16);
}
DEV float bf2f(unsigned short s) {
  unsigned int u = ((unsigned int)s) << 16;
  return __builtin_bit_cast(float, u);
}
DEV bf16_t bfraw(unsigned short s) { return __builtin_bit_cast(bf16_t, s); }

// ---------- fragment loaders (wave32 WMMA, contiguous k = 16*half + i in both A and B) ----------
// A 16x32 bf16: lane = M (lane&15); high half (lane>=16) owns K
// {16..31}, low half owns {0..15}: two contiguous 16B loads.
DEV void loadA16x32(const unsigned short* rowp, int half, v16bf& f) {
  const unsigned short* p = rowp + half * 16;
  union { uint4 u; __bf16 b[8]; } c0, c1;
  c0.u = *reinterpret_cast<const uint4*>(p);
  c1.u = *reinterpret_cast<const uint4*>(p + 8);
#pragma unroll
  for (int i = 0; i < 8; ++i) { f[i] = c0.b[i]; f[i + 8] = c1.b[i]; }
}
// B 32x16 bf16: lane = N (lane&15); element i -> K = (lane>=16?16:0)+i.
// One contiguous 32B load per lane.
DEV void loadB32x16(const unsigned short* p, v16bf& f) {
  union { uint4 u[2]; __bf16 b[16]; } c;
  c.u[0] = *reinterpret_cast<const uint4*>(p);
  c.u[1] = *reinterpret_cast<const uint4*>(p + 8);
#pragma unroll
  for (int i = 0; i < 16; ++i) f[i] = c.b[i];
}

// ---------- kernel: f32 -> bf16 convert ----------
__global__ void k_cvt_bf16(const float* __restrict__ in,
                           unsigned short* __restrict__ out, int n) {
  int i = blockIdx.x * blockDim.x + threadIdx.x;
  if (i < n) out[i] = f2bf(in[i]);
}

// ---------- kernel: LayerNorm row -> bf16 ----------
__global__ __launch_bounds__(256) void k_layernorm(
    const float* __restrict__ x, const float* __restrict__ g,
    const float* __restrict__ bta, unsigned short* __restrict__ xn) {
  __shared__ float ss[8], ss2[8];
  int row = blockIdx.x, tid = threadIdx.x;
  const float* xr = x + (size_t)row * DD;
  float s = 0.f, s2 = 0.f;
  for (int i = tid; i < DD; i += 256) { float v = xr[i]; s += v; s2 += v * v; }
#pragma unroll
  for (int m = 16; m >= 1; m >>= 1) { s += __shfl_xor(s, m, 32); s2 += __shfl_xor(s2, m, 32); }
  if ((tid & 31) == 0) { ss[tid >> 5] = s; ss2[tid >> 5] = s2; }
  __syncthreads();
  float ts = 0.f, ts2 = 0.f;
#pragma unroll
  for (int i = 0; i < 8; ++i) { ts += ss[i]; ts2 += ss2[i]; }
  float mu  = ts / DD;
  float var = ts2 / DD - mu * mu;
  float inv = rsqrtf(var + 1e-5f);
  for (int i = tid; i < DD; i += 256)
    xn[(size_t)row * DD + i] = f2bf((xr[i] - mu) * inv * g[i] + bta[i]);
}

// ---------- kernel: WMMA bf16 GEMM  C[M,N] = A[M,K] * W[N,K]^T ----------
// 256 threads = 8 waves (2 row-waves x 4 col-waves). Each wave computes a
// 2x2 grid of 16x16 C tiles: 4 WMMAs per K-step from 2 A + 2 B fragments
// (16 FLOP/byte operand intensity). Block tile = 64 rows x 128 cols.
__global__ __launch_bounds__(256) void k_gemm_bf16(
    const unsigned short* __restrict__ A, const unsigned short* __restrict__ W,
    const float* __restrict__ bias, float* __restrict__ outF,
    unsigned short* __restrict__ outB, int M, int N, int K) {
  int lane = threadIdx.x & 31;
  int w    = threadIdx.x >> 5;
  int half = lane >> 4;
  int hr   = lane & 15;
  int wr   = w & 1;          // row-wave
  int wc   = w >> 1;         // col-wave
  int row0 = blockIdx.y * 64 + wr * 32;    // this wave: rows row0..row0+31
  int col0 = blockIdx.x * 128 + wc * 32;   // this wave: cols col0..col0+31

  const unsigned short* arow0 = A + (size_t)(row0 + hr) * K;
  const unsigned short* arow1 = A + (size_t)(row0 + 16 + hr) * K;
  const unsigned short* wrow0 = W + (size_t)(col0 + hr) * K + half * 16;
  const unsigned short* wrow1 = W + (size_t)(col0 + 16 + hr) * K + half * 16;

  v8f acc00 = (v8f)0.0f, acc01 = (v8f)0.0f;
  v8f acc10 = (v8f)0.0f, acc11 = (v8f)0.0f;

  for (int kb = 0; kb < K; kb += 32) {
    __builtin_prefetch(arow0 + kb + 128, 0, 1);   // global_prefetch next tiles
    v16bf a0, a1, b0, b1;
    loadA16x32(arow0 + kb, half, a0);
    loadA16x32(arow1 + kb, half, a1);
    loadB32x16(wrow0 + kb, b0);
    loadB32x16(wrow1 + kb, b1);
    acc00 = __builtin_amdgcn_wmma_f32_16x16x32_bf16(false, a0, false, b0, (short)0, acc00, false, false);
    acc01 = __builtin_amdgcn_wmma_f32_16x16x32_bf16(false, a0, false, b1, (short)0, acc01, false, false);
    acc10 = __builtin_amdgcn_wmma_f32_16x16x32_bf16(false, a1, false, b0, (short)0, acc10, false, false);
    acc11 = __builtin_amdgcn_wmma_f32_16x16x32_bf16(false, a1, false, b1, (short)0, acc11, false, false);
  }

  v8f* accs[4] = { &acc00, &acc01, &acc10, &acc11 };
#pragma unroll
  for (int t = 0; t < 4; ++t) {
    int ri = t >> 1, cj = t & 1;
    v8f acc = *accs[t];
#pragma unroll
    for (int r = 0; r < 8; ++r) {
      int rr = row0 + ri * 16 + r + half * 8;
      int cc = col0 + cj * 16 + hr;
      float v = acc[r];
      if (bias) v += bias[cc];
      if (outF) outF[(size_t)rr * N + cc] = v;
      if (outB) outB[(size_t)rr * N + cc] = f2bf(v);
    }
  }
}

// ---------- kernel: RoPE + head transpose ----------
// qkv bf16 [B*N, 3072] -> qh/kh (rope applied), vh  each [B*H, N, 64] bf16
__global__ void k_rope(const unsigned short* __restrict__ qkv,
                       const float* __restrict__ rot,
                       unsigned short* __restrict__ qh,
                       unsigned short* __restrict__ kh,
                       unsigned short* __restrict__ vh) {
  int tid = blockIdx.x * blockDim.x + threadIdx.x;   // over B*N*H*64
  if (tid >= BB * NN * INNER) return;
  int dh = tid & 63;
  int h  = (tid >> 6) & (HH - 1);
  int n  = (tid >> 10) & (NN - 1);
  int b  = tid >> 20;
  size_t rowq = (size_t)(b * NN + n) * E3;
  int col  = h * 64 + dh;
  int pair = (dh < 32) ? dh + 32 : dh - 32;
  float sgn = (dh < 32) ? -1.f : 1.f;
  float e  = rot[((size_t)b * NN + n) * 64 + dh];
  float ce = cosf(e), se = sinf(e);
  float qv = bf2f(qkv[rowq + col]);
  float qp = bf2f(qkv[rowq + h * 64 + pair]);
  float kv = bf2f(qkv[rowq + INNER + col]);
  float kp = bf2f(qkv[rowq + INNER + h * 64 + pair]);
  float vv = bf2f(qkv[rowq + 2 * INNER + col]);
  size_t o = ((size_t)(b * HH + h) * NN + n) * 64 + dh;
  qh[o] = f2bf(qv * ce + sgn * qp * se);
  kh[o] = f2bf(kv * ce + sgn * kp * se);
  vh[o] = f2bf(vv);
}

// ---------- kernel: causal flash attention (one wave per 16-query tile) ----
__global__ __launch_bounds__(32) void k_attn(
    const unsigned short* __restrict__ qh, const unsigned short* __restrict__ kh,
    const unsigned short* __restrict__ vh, const unsigned char* __restrict__ xmask,
    unsigned short* __restrict__ outb) {
  __shared__ unsigned short ldsP[16 * 32];  // P tile staging (C-layout -> A-layout)
  int lane = threadIdx.x & 31;
  int half = lane >> 4;
  int hr   = lane & 15;
  int bh = blockIdx.x;
  int b  = bh >> 4;
  int h  = bh & (HH - 1);
  int q0 = blockIdx.y * 16;
  const unsigned short* qp = qh + (size_t)bh * NN * 64;
  const unsigned short* kp = kh + (size_t)bh * NN * 64;
  const unsigned short* vp = vh + (size_t)bh * NN * 64;
  const unsigned char*  mk = xmask + (size_t)b * NN;

  v16bf aq0, aq1;                                  // Q fragment, dims 0-31 / 32-63
  loadA16x32(qp + (size_t)(q0 + hr) * 64, half, aq0);
  loadA16x32(qp + (size_t)(q0 + hr) * 64 + 32, half, aq1);

  float mi[8], li[8];
  v8f oacc[4];
#pragma unroll
  for (int r = 0; r < 8; ++r) { mi[r] = -INFINITY; li[r] = 0.f; }
#pragma unroll
  for (int c = 0; c < 4; ++c) oacc[c] = (v8f)0.0f;

  int iters = (q0 + 16 + 31) >> 5;                 // 32 keys per iteration
  for (int it = 0; it < iters; ++it) {
    int key0 = it * 32;
    // ---- S = Q * K^T for two 16-key tiles ----
    v16bf bk;
    v8f s0 = (v8f)0.0f, s1 = (v8f)0.0f;
    loadB32x16(kp + (size_t)(key0 + hr) * 64 + half * 16, bk);
    s0 = __builtin_amdgcn_wmma_f32_16x16x32_bf16(false, aq0, false, bk, (short)0, s0, false, false);
    loadB32x16(kp + (size_t)(key0 + hr) * 64 + 32 + half * 16, bk);
    s0 = __builtin_amdgcn_wmma_f32_16x16x32_bf16(false, aq1, false, bk, (short)0, s0, false, false);
    loadB32x16(kp + (size_t)(key0 + 16 + hr) * 64 + half * 16, bk);
    s1 = __builtin_amdgcn_wmma_f32_16x16x32_bf16(false, aq0, false, bk, (short)0, s1, false, false);
    loadB32x16(kp + (size_t)(key0 + 16 + hr) * 64 + 32 + half * 16, bk);
    s1 = __builtin_amdgcn_wmma_f32_16x16x32_bf16(false, aq1, false, bk, (short)0, s1, false, false);

    // ---- scale + causal/pad mask, online softmax update ----
    float p0[8], p1[8];
#pragma unroll
    for (int r = 0; r < 8; ++r) {
      int qrow = q0 + r + half * 8;
      int n0 = key0 + hr, n1 = key0 + 16 + hr;
      float v0 = s0[r] * 0.125f, v1 = s1[r] * 0.125f;
      if (n0 > qrow || !mk[n0]) v0 = -INFINITY;
      if (n1 > qrow || !mk[n1]) v1 = -INFINITY;
      p0[r] = v0; p1[r] = v1;
    }
#pragma unroll
    for (int r = 0; r < 8; ++r) {
      float bm = fmaxf(p0[r], p1[r]);
#pragma unroll
      for (int m = 8; m >= 1; m >>= 1) bm = fmaxf(bm, __shfl_xor(bm, m, 32));
      float mnew  = fmaxf(mi[r], bm);
      float alpha, e0, e1;
      if (mnew == -INFINITY) { alpha = 1.f; e0 = 0.f; e1 = 0.f; }
      else {
        alpha = (mi[r] == -INFINITY) ? 0.f : __expf(mi[r] - mnew);
        e0 = (p0[r] == -INFINITY) ? 0.f : __expf(p0[r] - mnew);
        e1 = (p1[r] == -INFINITY) ? 0.f : __expf(p1[r] - mnew);
        mi[r] = mnew;
      }
      float rs = e0 + e1;
#pragma unroll
      for (int m = 8; m >= 1; m >>= 1) rs += __shfl_xor(rs, m, 32);
      li[r] = li[r] * alpha + rs;
      p0[r] = e0; p1[r] = e1;
#pragma unroll
      for (int c = 0; c < 4; ++c) oacc[c][r] *= alpha;
    }

    // ---- P (C-layout f32) -> LDS -> A-layout bf16 fragment ----
    __syncthreads();
#pragma unroll
    for (int r = 0; r < 8; ++r) {
      int row = r + half * 8;
      ldsP[row * 32 + hr]      = f2bf(p0[r]);
      ldsP[row * 32 + 16 + hr] = f2bf(p1[r]);
    }
    __syncthreads();
    v16bf ap;
#pragma unroll
    for (int i = 0; i < 16; ++i) {
      int j = i >> 1, hb = i & 1;
      int k = half * 16 + (j << 1) + hb;
      ap[i] = bfraw(ldsP[hr * 32 + k]);
    }

    // ---- O += P * V ----
#pragma unroll
    for (int c = 0; c < 4; ++c) {
      v16bf bv;
#pragma unroll
      for (int i = 0; i < 16; ++i) {
        int key = key0 + half * 16 + i;
        bv[i] = bfraw(vp[(size_t)key * 64 + c * 16 + hr]);
      }
      oacc[c] = __builtin_amdgcn_wmma_f32_16x16x32_bf16(false, ap, false, bv,
                                                        (short)0, oacc[c], false, false);
    }
  }

  // ---- normalize + write back to token-major [B*N, INNER] bf16 ----
#pragma unroll
  for (int r = 0; r < 8; ++r) {
    int qrow = q0 + r + half * 8;
    float inv = (li[r] > 0.f) ? 1.f / li[r] : 0.f;
    size_t orow = ((size_t)(b * NN + qrow)) * INNER + h * 64;
#pragma unroll
    for (int c = 0; c < 4; ++c)
      outb[orow + c * 16 + hr] = f2bf(oacc[c][r] * inv);
  }
}

// ---------- host launcher ----------
extern "C" void kernel_launch(void* const* d_in, const int* in_sizes, int n_in,
                              void* d_out, int out_size, void* d_ws, size_t ws_size,
                              hipStream_t stream) {
  const float* x    = (const float*)d_in[0];
  const float* rot  = (const float*)d_in[1];
  const unsigned char* xmask = (const unsigned char*)d_in[2];
  const float* lng  = (const float*)d_in[3];
  const float* lnb  = (const float*)d_in[4];
  const float* wqkv = (const float*)d_in[5];
  const float* wout = (const float*)d_in[6];
  const float* bout = (const float*)d_in[7];
  float* out = (float*)d_out;

  char* ws = (char*)d_ws;
  size_t off = 0;
  auto alloc = [&](size_t bytes) { char* p = ws + off; off += (bytes + 255) & ~(size_t)255; return p; };
  unsigned short* xn_bf   = (unsigned short*)alloc((size_t)MROWS * DD * 2);      // 8 MiB
  unsigned short* wqkv_bf = (unsigned short*)alloc((size_t)E3 * DD * 2);         // 6 MiB
  unsigned short* wout_bf = (unsigned short*)alloc((size_t)DD * INNER * 2);      // 2 MiB
  unsigned short* qkv_bf  = (unsigned short*)alloc((size_t)MROWS * E3 * 2);      // 24 MiB
  unsigned short* qh      = (unsigned short*)alloc((size_t)BB * HH * NN * DH * 2);
  unsigned short* kh      = (unsigned short*)alloc((size_t)BB * HH * NN * DH * 2);
  unsigned short* vh      = (unsigned short*)alloc((size_t)BB * HH * NN * DH * 2);
  unsigned short* ao_bf   = (unsigned short*)alloc((size_t)MROWS * INNER * 2);   // 8 MiB
  (void)ws_size; (void)in_sizes; (void)n_in; (void)out_size;

  // 1) weights -> bf16
  {
    int n = E3 * DD;
    k_cvt_bf16<<<(n + 255) / 256, 256, 0, stream>>>(wqkv, wqkv_bf, n);
    n = DD * INNER;
    k_cvt_bf16<<<(n + 255) / 256, 256, 0, stream>>>(wout, wout_bf, n);
  }
  // 2) LayerNorm -> bf16
  k_layernorm<<<MROWS, 256, 0, stream>>>(x, lng, lnb, xn_bf);
  // 3) QKV GEMM (WMMA bf16, f32 accum, 2x2 register-blocked) -> bf16
  k_gemm_bf16<<<dim3(E3 / 128, MROWS / 64), 256, 0, stream>>>(
      xn_bf, wqkv_bf, nullptr, nullptr, qkv_bf, MROWS, E3, DD);
  // 4) RoPE + head transpose
  {
    int n = BB * NN * INNER;
    k_rope<<<(n + 255) / 256, 256, 0, stream>>>(qkv_bf, rot, qh, kh, vh);
  }
  // 5) causal flash attention
  k_attn<<<dim3(BB * HH, NN / 16), 32, 0, stream>>>(qh, kh, vh, xmask, ao_bf);
  // 6) output GEMM + bias -> f32 d_out (2x2 register-blocked)
  k_gemm_bf16<<<dim3(DD / 128, MROWS / 64), 256, 0, stream>>>(
      ao_bf, wout_bf, bout, out, nullptr, MROWS, DD, INNER);
}
